// Residual_Conv_Mamba_463856468132
// MI455X (gfx1250) — matrix operand order repaired and checked by reading, not compiled, not run on hardware
//
#include <hip/hip_runtime.h>
#include <cstdint>
#include <cstddef>

typedef __attribute__((ext_vector_type(16))) _Float16 v16h;
typedef __attribute__((ext_vector_type(8)))  _Float16 v8h;
typedef __attribute__((ext_vector_type(8)))  float    v8f;

// =====================================================================
// WMMA GEMM:  C(MxN) = A(MxKp) * B(NxKp)^T  (+bias per col, +softplus,
// +residual).  f16 inputs, f32 accumulate via v_wmma_f32_16x16x32_f16.
// One wave computes a 16(M) x 16*NT(N) register tile; NT is a template
// parameter so the K-loop is branch-free. Per K-step the A fragment and
// all NT B fragments are loaded into distinct registers FIRST, then the
// NT WMMAs are issued: this lets the scheduler clause the loads and use
// partial s_wait_loadcnt so WMMA j overlaps loads j+1.. (latency hiding).
// Kp must be a multiple of 32; rows 16B-aligned.
// =====================================================================
template <int NT>
__global__ void __launch_bounds__(32)
gemm_wmma_k(const _Float16* __restrict__ A, const _Float16* __restrict__ B,
            const float* __restrict__ bias, const float* __restrict__ residual,
            float* __restrict__ C, int M, int N, int Kp, int act, int n_base)
{
  const int n0   = n_base + (blockIdx.x << 6); // 64 cols per full-tile wave
  const int m0   = blockIdx.y << 4;            // 16 rows per wave
  const int lane = threadIdx.x;
  const int r    = lane & 15;
  const int hi   = lane >> 4;

  int arow = m0 + r; if (arow > M - 1) arow = M - 1;
  // A fragment (16x32 f16): lane holds row m0+r; elems 0..7 -> K=k0+hi*8..,
  // elems 8..15 -> K=k0+16+hi*8..  (ISA 7.12.2, 16-bit A 16x32)
  const _Float16* ap = A + (size_t)arow * Kp + (hi << 3);

  // B fragments (32x16 f16): lane holds col n0+j*16+r; elems 0..7 -> K=k0+hi*8..,
  // elems 8..15 -> K=k0+16+hi*8.. (same order as A)
  const _Float16* bp[NT];
#pragma unroll
  for (int j = 0; j < NT; ++j) {
    int brow = n0 + (j << 4) + r; if (brow > N - 1) brow = N - 1;
    bp[j] = B + (size_t)brow * Kp + (hi << 3);
  }

  v8f acc[NT];
#pragma unroll
  for (int j = 0; j < NT; ++j) acc[j] = (v8f){0.f,0.f,0.f,0.f,0.f,0.f,0.f,0.f};

  union U { v16h v; v8h h[2]; };
  for (int k0 = 0; k0 < Kp; k0 += 32) {
    U ua, ub[NT];
    ua.h[0] = *(const v8h*)(ap + k0);
    ua.h[1] = *(const v8h*)(ap + k0 + 16);
#pragma unroll
    for (int j = 0; j < NT; ++j) {
      ub[j].h[0] = *(const v8h*)(bp[j] + k0);
      ub[j].h[1] = *(const v8h*)(bp[j] + k0 + 16);
    }
#pragma unroll
    for (int j = 0; j < NT; ++j)
      acc[j] = __builtin_amdgcn_wmma_f32_16x16x32_f16(false, ua.v, false, ub[j].v,
                                                      (short)0, acc[j], false, false);
  }

#pragma unroll
  for (int j = 0; j < NT; ++j) {
    const int n = n0 + (j << 4) + r;
    if (n < N) {
      const float bv = bias ? bias[n] : 0.f;
#pragma unroll
      for (int v = 0; v < 8; ++v) {
        const int m = m0 + v + (hi << 3);    // C/D layout: VGPR v -> row v+8*hi
        if (m < M) {
          float val = acc[j][v] + bv;
          if (act == 1) val = (val > 20.f) ? val : log1pf(__expf(val)); // softplus
          if (residual) val += residual[(size_t)m * N + n];
          C[(size_t)m * N + n] = val;
        }
      }
    }
  }
}

// ---------------- im2col (f32 src -> f16 col, K zero-padded) -----------
// dst[(b*Lout+t)*Kp + ci*KW + j] = src(b, ci, t*stride - pad + j)
// chmajor=1: src is (B,C,L); chmajor=0: src is (B,L,C)
__global__ void im2col_k(const float* __restrict__ src, _Float16* __restrict__ dst,
                         int Bn, int Lin, int Cin, int Lout, int KW, int stride,
                         int pad, int Kp, int chmajor)
{
  size_t idx = (size_t)blockIdx.x * 256 + threadIdx.x;
  size_t total = (size_t)Bn * Lout * Kp;
  if (idx >= total) return;
  int kp = (int)(idx % Kp);
  size_t m = idx / Kp;
  int t = (int)(m % Lout);
  int b = (int)(m / Lout);
  float v = 0.f;
  if (kp < Cin * KW) {
    int ci = kp / KW, j = kp % KW;
    int pos = t * stride - pad + j;
    if (pos >= 0 && pos < Lin) {
      v = chmajor ? src[((size_t)b * Cin + ci) * Lin + pos]
                  : src[((size_t)b * Lin + pos) * Cin + ci];
    }
  }
  dst[idx] = (_Float16)v;
}

// ---------------- f32 -> f16 with K padding (weights etc.) -------------
__global__ void f16pad_k(const float* __restrict__ src, _Float16* __restrict__ dst,
                         int rows, int K, int Kp)
{
  size_t idx = (size_t)blockIdx.x * 256 + threadIdx.x;
  size_t total = (size_t)rows * Kp;
  if (idx >= total) return;
  int k = (int)(idx % Kp);
  size_t r = idx / Kp;
  dst[idx] = (_Float16)((k < K) ? src[r * K + k] : 0.f);
}

// ---------------- BatchNorm batch statistics (per channel) -------------
__global__ void __launch_bounds__(256)
bnstats_k(const float* __restrict__ x, int M, int C,
          float* __restrict__ mean, float* __restrict__ var)
{
  int c = blockIdx.x, tid = threadIdx.x;
  float s = 0.f, s2 = 0.f;
  for (int m = tid; m < M; m += 256) {
    float v = x[(size_t)m * C + c];
    s += v; s2 += v * v;
  }
  __shared__ float rs[256], rq[256];
  rs[tid] = s; rq[tid] = s2; __syncthreads();
  for (int off = 128; off > 0; off >>= 1) {
    if (tid < off) { rs[tid] += rs[tid + off]; rq[tid] += rq[tid + off]; }
    __syncthreads();
  }
  if (tid == 0) {
    float mu = rs[0] / (float)M;
    mean[c] = mu;
    var[c]  = rq[0] / (float)M - mu * mu;
  }
}

// ---------------- BN apply (+residual, +relu) --------------------------
__global__ void bnapply_k(const float* __restrict__ x, const float* __restrict__ mean,
                          const float* __restrict__ var, const float* __restrict__ g,
                          const float* __restrict__ b, const float* __restrict__ res,
                          float* __restrict__ out, size_t M, int C, int relu)
{
  size_t idx = (size_t)blockIdx.x * 256 + threadIdx.x;
  size_t total = M * (size_t)C;
  if (idx >= total) return;
  int c = (int)(idx % C);
  float v = (x[idx] - mean[c]) * rsqrtf(var[c] + 1e-5f) * g[c] + b[c];
  if (res) v += res[idx];
  if (relu) v = fmaxf(v, 0.f);
  out[idx] = v;
}

// ---------------- maxpool3 stride2 pad1 over t, layout (B,L,C) ---------
__global__ void maxpool_k(const float* __restrict__ x, float* __restrict__ out,
                          int Bn, int Lin, int C, int Lout)
{
  size_t idx = (size_t)blockIdx.x * 256 + threadIdx.x;
  size_t total = (size_t)Bn * Lout * C;
  if (idx >= total) return;
  int c = (int)(idx % C);
  size_t tb = idx / C;
  int t = (int)(tb % Lout);
  int b = (int)(tb / Lout);
  float m = -3.4e38f;
  for (int j = 0; j < 3; ++j) {
    int p = 2 * t - 1 + j;
    if (p >= 0 && p < Lin) m = fmaxf(m, x[((size_t)b * Lin + p) * C + c]);
  }
  out[idx] = m;
}

// ---------------- build embed A operand: (b,1875,128) -> (4096,1888)f16 -
__global__ void embedA_k(const float* __restrict__ src, _Float16* __restrict__ dst)
{
  size_t idx = (size_t)blockIdx.x * 256 + threadIdx.x;
  if (idx >= (size_t)4096 * 1888) return;
  int k = (int)(idx % 1888);
  int m = (int)(idx / 1888);
  int co = m & 127, b = m >> 7;
  float v = (k < 1875) ? src[((size_t)b * 1875 + k) * 128 + co] : 0.f;
  dst[idx] = (_Float16)v;
}

// ---------------- RMSNorm over last dim D (row per block) --------------
__global__ void __launch_bounds__(256)
rms_k(const float* __restrict__ x, const float* __restrict__ w,
      float* __restrict__ outf, _Float16* __restrict__ outh, int D)
{
  int row = blockIdx.x, tid = threadIdx.x;
  const float* xr = x + (size_t)row * D;
  float s = 0.f;
  for (int i = tid; i < D; i += 256) { float v = xr[i]; s += v * v; }
  __shared__ float red[256];
  red[tid] = s; __syncthreads();
  for (int off = 128; off > 0; off >>= 1) {
    if (tid < off) red[tid] += red[tid + off];
    __syncthreads();
  }
  float rs = rsqrtf(red[0] / (float)D + 1e-5f);
  for (int i = tid; i < D; i += 256) {
    float v = xr[i] * rs * w[i];
    if (outf) outf[(size_t)row * D + i] = v;
    if (outh) outh[(size_t)row * D + i] = (_Float16)v;
  }
}

// ---------------- depthwise causal conv4 + bias + silu -----------------
// xr:(4096,2048) use cols [0,1024); out f32 + f16 (4096,1024)
__global__ void dwconv_k(const float* __restrict__ xr, const float* __restrict__ w,
                         const float* __restrict__ bias, float* __restrict__ outf,
                         _Float16* __restrict__ outh)
{
  size_t idx = (size_t)blockIdx.x * 256 + threadIdx.x;
  if (idx >= (size_t)4096 * 1024) return;
  int d = (int)(idx & 1023);
  size_t m = idx >> 10;
  int l = (int)(m & 127);
  size_t b = m >> 7;
  float acc = bias[d];
#pragma unroll
  for (int j = 0; j < 4; ++j) {
    int pos = l - 3 + j;
    if (pos >= 0) acc += w[d * 4 + j] * xr[((b * 128 + (size_t)pos) * 2048) + d];
  }
  float s = acc / (1.f + __expf(-acc));   // silu
  outf[idx] = s;
  outh[idx] = (_Float16)s;
}

// ---------------- slice first 32 cols of xdbl (delta raw) to f16 -------
__global__ void slice32_k(const float* __restrict__ xdbl, _Float16* __restrict__ dst)
{
  size_t idx = (size_t)blockIdx.x * 256 + threadIdx.x;
  if (idx >= (size_t)4096 * 32) return;
  int k = (int)(idx & 31);
  size_t m = idx >> 5;
  dst[idx] = (_Float16)xdbl[m * 64 + k];
}

// ---------------- selective scan + D skip + silu(res) gate -> f16 ------
// thread per (b,d); 16 SSM states in registers, t loop = 128
__global__ void __launch_bounds__(256)
scan_k(const float* __restrict__ u,      // (4096,1024) conv+silu out
       const float* __restrict__ delta,  // (4096,1024) softplus'd
       const float* __restrict__ xdbl,   // (4096,64): B at +32, C at +48
       const float* __restrict__ A_log,  // (1024,16)
       const float* __restrict__ Dp,     // (1024)
       const float* __restrict__ xr,     // (4096,2048): res at +1024
       _Float16* __restrict__ yh)        // (4096,1024) gated output (f16)
{
  int idx = blockIdx.x * 256 + threadIdx.x;   // 32768
  int d = idx & 1023;
  int b = idx >> 10;
  float Arow[16], h[16];
#pragma unroll
  for (int n = 0; n < 16; ++n) { Arow[n] = -__expf(A_log[d * 16 + n]); h[n] = 0.f; }
  float Dd = Dp[d];
  for (int t = 0; t < 128; ++t) {
    size_t m = (size_t)b * 128 + t;
    float dl = delta[m * 1024 + d];
    float uv = u[m * 1024 + d];
    float du = dl * uv;
    const float* bc = xdbl + m * 64;
    float acc = 0.f;
#pragma unroll
    for (int n = 0; n < 16; ++n) {
      float dA = __expf(dl * Arow[n]);
      h[n] = dA * h[n] + du * bc[32 + n];
      acc += h[n] * bc[48 + n];
    }
    float y = acc + uv * Dd;
    float r = xr[m * 2048 + 1024 + d];
    y *= r / (1.f + __expf(-r));           // * silu(res)
    yh[m * 1024 + d] = (_Float16)y;
  }
}

// ---------------- mean over l=128 -> f16 (32,512) ----------------------
__global__ void meanpool_k(const float* __restrict__ hn, _Float16* __restrict__ outh)
{
  int idx = blockIdx.x * 256 + threadIdx.x;
  if (idx >= 32 * 512) return;
  int f = idx & 511, b = idx >> 9;
  float s = 0.f;
  for (int l = 0; l < 128; ++l) s += hn[((size_t)b * 128 + l) * 512 + f];
  outh[idx] = (_Float16)(s * (1.f / 128.f));
}

// =====================================================================
extern "C" void kernel_launch(void* const* d_in, const int* in_sizes, int n_in,
                              void* d_out, int out_size, void* d_ws, size_t ws_size,
                              hipStream_t stream)
{
  (void)in_sizes; (void)n_in; (void)out_size; (void)ws_size;

  // ---- inputs (recursive dict-insertion order of setup_inputs) --------
  const float* X      = (const float*)d_in[0];
  const float* stem_w = (const float*)d_in[1];
  const float* stem_g = (const float*)d_in[2];
  const float* stem_b = (const float*)d_in[3];
  const float* bw[4][9];
  int bi = 4;
  for (int i = 0; i < 4; ++i) {
    for (int j = 0; j < 6; ++j) bw[i][j] = (const float*)d_in[bi++];
    if (i == 2) { for (int j = 6; j < 9; ++j) bw[i][j] = (const float*)d_in[bi++]; }
    else        { bw[i][6] = bw[i][7] = bw[i][8] = nullptr; }
  }
  const float* embed_w = (const float*)d_in[bi++];  // 31
  const float* embed_b = (const float*)d_in[bi++];  // 32
  const float* mp[4][10];
  for (int l = 0; l < 4; ++l)
    for (int j = 0; j < 10; ++j) mp[l][j] = (const float*)d_in[bi++];
  const float* fnorm_w = (const float*)d_in[bi++];  // 73
  const float* fc_w    = (const float*)d_in[bi++];  // 74
  float* OUT = (float*)d_out;

  // ---- workspace layout (256B-aligned bump allocator) -----------------
  char* base = (char*)d_ws; size_t off = 0;
  auto alloc = [&](size_t bytes) -> char* {
    char* p = base + off; off += (bytes + 255) & ~(size_t)255; return p;
  };
  _Float16* col16 = (_Float16*)alloc(53760000ull * 2);   // max im2col
  _Float16* w16   = (_Float16*)alloc(1048576ull * 2);    // max weight f16
  _Float16* a16   = (_Float16*)alloc(7733248ull * 2);    // 4096x1888 f16
  float* stemout  = (float*)alloc(15360000ull * 4);      // 240000x64
  float* B0 = (float*)alloc(7680000ull * 4);
  float* B1 = (float*)alloc(7680000ull * 4);
  float* B2 = (float*)alloc(7680000ull * 4);
  float* B3 = (float*)alloc(7680000ull * 4);
  float* d_mean = (float*)alloc(128 * 4);
  float* d_var  = (float*)alloc(128 * 4);
  float* h     = (float*)alloc(2097152ull * 4);          // 4096x512
  float* xr    = (float*)alloc(8388608ull * 4);          // 4096x2048
  float* xmc   = (float*)alloc(4194304ull * 4);          // 4096x1024
  float* xdbl  = (float*)alloc(262144ull  * 4);          // 4096x64
  float* delta = (float*)alloc(4194304ull * 4);          // 4096x1024
  float* hn    = (float*)alloc(2097152ull * 4);          // 4096x512

  // ---- GEMM dispatch: full 64-col tiles (NT=4) + column tail ----------
  auto launch_gemm = [&](const _Float16* A, const _Float16* Bw, const float* bias,
                         const float* res, float* Cp, int Mr, int Nn, int Kp,
                         int act) {
    unsigned mt = (unsigned)((Mr + 15) / 16);
    int nfull = Nn >> 6;
    if (nfull > 0)
      gemm_wmma_k<4><<<dim3((unsigned)nfull, mt), dim3(32), 0, stream>>>(
          A, Bw, bias, res, Cp, Mr, Nn, Kp, act, 0);
    int rem = Nn - (nfull << 6);
    if (rem > 0) {
      int nb = nfull << 6;
      int nt = (rem + 15) / 16;
      if (nt == 1)
        gemm_wmma_k<1><<<dim3(1, mt), dim3(32), 0, stream>>>(
            A, Bw, bias, res, Cp, Mr, Nn, Kp, act, nb);
      else if (nt == 2)
        gemm_wmma_k<2><<<dim3(1, mt), dim3(32), 0, stream>>>(
            A, Bw, bias, res, Cp, Mr, Nn, Kp, act, nb);
      else
        gemm_wmma_k<3><<<dim3(1, mt), dim3(32), 0, stream>>>(
            A, Bw, bias, res, Cp, Mr, Nn, Kp, act, nb);
    }
  };

  // ---- helpers --------------------------------------------------------
  auto convgemm = [&](const float* src, int Bn, int Lin, int Cin, int chmajor,
                      const float* w, int Cout, int KW, int stride, int pad,
                      int Lout, float* out) {
    int Kred = Cin * KW, Kp = (Kred + 31) & ~31;
    size_t Mrows = (size_t)Bn * Lout;
    size_t tot = Mrows * (size_t)Kp;
    im2col_k<<<dim3((unsigned)((tot + 255) / 256)), dim3(256), 0, stream>>>(
        src, col16, Bn, Lin, Cin, Lout, KW, stride, pad, Kp, chmajor);
    size_t wt = (size_t)Cout * Kp;
    f16pad_k<<<dim3((unsigned)((wt + 255) / 256)), dim3(256), 0, stream>>>(
        w, w16, Cout, Kred, Kp);
    launch_gemm(col16, w16, nullptr, nullptr, out, (int)Mrows, Cout, Kp, 0);
  };
  auto gemm = [&](const _Float16* A, const float* W, int Mr, int Nn, int K,
                  const float* bias, const float* res, float* Cout, int act) {
    int Kp = (K + 31) & ~31;
    size_t wt = (size_t)Nn * Kp;
    f16pad_k<<<dim3((unsigned)((wt + 255) / 256)), dim3(256), 0, stream>>>(
        W, w16, Nn, K, Kp);
    launch_gemm(A, w16, bias, res, Cout, Mr, Nn, Kp, act);
  };
  auto bn = [&](float* x, size_t Mrows, int C, const float* g, const float* b,
                const float* res, int relu, float* out) {
    bnstats_k<<<dim3(C), dim3(256), 0, stream>>>(x, (int)Mrows, C, d_mean, d_var);
    size_t tot = Mrows * (size_t)C;
    bnapply_k<<<dim3((unsigned)((tot + 255) / 256)), dim3(256), 0, stream>>>(
        x, d_mean, d_var, g, b, res, out, Mrows, C, relu);
  };

  // ---- stem: conv(12->64,k15,s2,p7) + BN + relu + maxpool -------------
  convgemm(X, 32, 15000, 12, 1, stem_w, 64, 15, 2, 7, 7500, stemout);
  bn(stemout, 240000, 64, stem_g, stem_b, nullptr, 1, stemout);
  maxpool_k<<<dim3((7680000 + 255) / 256), dim3(256), 0, stream>>>(
      stemout, B0, 32, 7500, 64, 3750);

  // ---- basic blocks ---------------------------------------------------
  // block0 (64->64, s1): in B0
  convgemm(B0, 32, 3750, 64, 0, bw[0][0], 64, 7, 1, 3, 3750, B1);
  bn(B1, 120000, 64, bw[0][1], bw[0][2], nullptr, 1, B1);
  convgemm(B1, 32, 3750, 64, 0, bw[0][3], 64, 7, 1, 3, 3750, B2);
  bn(B2, 120000, 64, bw[0][4], bw[0][5], B0, 1, B2);
  // block1 (64->64, s1): in B2
  convgemm(B2, 32, 3750, 64, 0, bw[1][0], 64, 7, 1, 3, 3750, B0);
  bn(B0, 120000, 64, bw[1][1], bw[1][2], nullptr, 1, B0);
  convgemm(B0, 32, 3750, 64, 0, bw[1][3], 64, 7, 1, 3, 3750, B1);
  bn(B1, 120000, 64, bw[1][4], bw[1][5], B2, 1, B1);
  // block2 (64->128, s2, downsample): in B1
  convgemm(B1, 32, 3750, 64, 0, bw[2][0], 128, 7, 2, 3, 1875, B0);
  bn(B0, 60000, 128, bw[2][1], bw[2][2], nullptr, 1, B0);
  convgemm(B0, 32, 1875, 128, 0, bw[2][3], 128, 7, 1, 3, 1875, B2);
  convgemm(B1, 32, 3750, 64, 0, bw[2][6], 128, 1, 2, 0, 1875, B3);  // 1x1 s2
  bn(B3, 60000, 128, bw[2][7], bw[2][8], nullptr, 0, B3);
  bn(B2, 60000, 128, bw[2][4], bw[2][5], B3, 1, B2);
  // block3 (128->128, s1): in B2
  convgemm(B2, 32, 1875, 128, 0, bw[3][0], 128, 7, 1, 3, 1875, B0);
  bn(B0, 60000, 128, bw[3][1], bw[3][2], nullptr, 1, B0);
  convgemm(B0, 32, 1875, 128, 0, bw[3][3], 128, 7, 1, 3, 1875, B1);
  bn(B1, 60000, 128, bw[3][4], bw[3][5], B2, 1, B1);

  // ---- embed: (32,128,1875) @ embed_w^T + b -> h (4096,512) -----------
  embedA_k<<<dim3((unsigned)((7733248ull + 255) / 256)), dim3(256), 0, stream>>>(
      B1, a16);
  gemm(a16, embed_w, 4096, 512, 1875, embed_b, nullptr, h, 0);

  // ---- mamba layers ---------------------------------------------------
  for (int l = 0; l < 4; ++l) {
    const float* norm_w    = mp[l][0];
    const float* in_proj_w = mp[l][1];
    const float* conv_w    = mp[l][2];
    const float* conv_b    = mp[l][3];
    const float* x_proj_w  = mp[l][4];
    const float* dt_proj_w = mp[l][5];
    const float* dt_proj_b = mp[l][6];
    const float* A_log     = mp[l][7];
    const float* Dp        = mp[l][8];
    const float* out_proj_w= mp[l][9];

    rms_k<<<dim3(4096), dim3(256), 0, stream>>>(h, norm_w, (float*)nullptr, a16, 512);
    gemm(a16, in_proj_w, 4096, 2048, 512, nullptr, nullptr, xr, 0);
    dwconv_k<<<dim3(4194304 / 256), dim3(256), 0, stream>>>(xr, conv_w, conv_b,
                                                            xmc, a16);
    gemm(a16, x_proj_w, 4096, 64, 1024, nullptr, nullptr, xdbl, 0);
    slice32_k<<<dim3(131072 / 256), dim3(256), 0, stream>>>(xdbl, a16);
    gemm(a16, dt_proj_w, 4096, 1024, 32, dt_proj_b, nullptr, delta, 1); // softplus
    scan_k<<<dim3(32768 / 256), dim3(256), 0, stream>>>(xmc, delta, xdbl, A_log,
                                                        Dp, xr, a16);
    gemm(a16, out_proj_w, 4096, 512, 1024, nullptr, h, h, 0);           // +residual
  }

  // ---- head: rms -> mean over l -> fc ---------------------------------
  rms_k<<<dim3(4096), dim3(256), 0, stream>>>(h, fnorm_w, hn, (_Float16*)nullptr, 512);
  meanpool_k<<<dim3((16384 + 255) / 256), dim3(256), 0, stream>>>(hn, a16);
  gemm(a16, fc_w, 32, 9, 512, nullptr, nullptr, OUT, 0);
}
